// TransformerLanguageModel_18734647345516
// MI455X (gfx1250) — matrix operand order repaired and checked by reading, not compiled, not run on hardware
//
#include <hip/hip_runtime.h>

// ---------------------------------------------------------------- types ----
typedef __bf16 bf16_t;
typedef __attribute__((ext_vector_type(8)))  __bf16 bf16x8;
typedef __attribute__((ext_vector_type(16))) __bf16 bf16x16;
typedef __attribute__((ext_vector_type(8)))  float  f32x8;
typedef __attribute__((ext_vector_type(4)))  unsigned int v4u;
typedef __attribute__((ext_vector_type(4)))  int v4i;
typedef __attribute__((ext_vector_type(8)))  int v8i;

// ------------------------------------------------------------- constants ---
constexpr int Lc = 8;
constexpr int Dm = 1024;     // model dim
constexpr int Hn = 16;       // heads
constexpr int DHc = 64;      // head dim
constexpr int Sc = 1024;     // seq len
constexpr int Bc = 4;        // batch
constexpr int Vc = 32000;    // vocab
constexpr int Mrows = Bc * Sc;   // 4096 token rows
constexpr int FF = 4 * Dm;       // 4096

// ------------------------------------------------------ WMMA primitives ----
__device__ __forceinline__ f32x8 wmma_bf16(bf16x16 a, bf16x16 b, f32x8 c) {
  // D = A(16x32) * B(32x16) + C ; f32 accumulate
  return __builtin_amdgcn_wmma_f32_16x16x32_bf16(
      /*neg_a=*/false, a, /*neg_b=*/false, b,
      /*c_mod=*/(short)0, c, /*reuse_a=*/false, /*reuse_b=*/false);
}

__device__ __forceinline__ bf16x16 ld_frag_pair(const bf16_t* p0, const bf16_t* p1) {
  bf16x8 lo = *(const bf16x8*)p0;
  bf16x8 hi = *(const bf16x8*)p1;
  return __builtin_shufflevector(lo, hi, 0,1,2,3,4,5,6,7,8,9,10,11,12,13,14,15);
}
__device__ __forceinline__ bf16x16 ld_frag_contig(const bf16_t* p) {
  return ld_frag_pair(p, p + 8);
}

// -------------------------------------------- CDNA5 async copy to LDS ------
// GLOBAL_LOAD_ASYNC_TO_LDS_B128: 16B/lane memory->LDS, no VGPR bounce,
// tracked with ASYNCcnt (CDNA5 ISA).
__device__ __forceinline__ void async_copy_b128(unsigned lds_addr, const void* gaddr) {
  asm volatile("global_load_async_to_lds_b128 %0, %1, off"
               :: "v"(lds_addr), "v"(gaddr)
               : "memory");
}
__device__ __forceinline__ void wait_async_le1() {
  asm volatile("s_wait_asynccnt 0x1" ::: "memory");
}
__device__ __forceinline__ void wait_async_le3() {
  asm volatile("s_wait_asynccnt 0x3" ::: "memory");
}
__device__ __forceinline__ void wait_async_0() {
  asm volatile("s_wait_asynccnt 0x0" ::: "memory");
}
// LDS byte address of a shared-memory object (low 32 bits of the pointer)
__device__ __forceinline__ unsigned lds_addr_of(const void* p) {
  return (unsigned)(uintptr_t)p;
}

// -------------------------------------- CDNA5 Tensor Data Mover (TDM) ------
#if __has_builtin(__builtin_amdgcn_tensor_load_to_lds)
#define USE_TDM 1
#else
#define USE_TDM 0
#endif

__device__ __forceinline__ void wait_tensor_le1() {
#if __has_builtin(__builtin_amdgcn_s_wait_tensorcnt)
  __builtin_amdgcn_s_wait_tensorcnt(1);
#else
  asm volatile("s_wait_tensorcnt 0x1" ::: "memory");
#endif
}
__device__ __forceinline__ void wait_tensor_0() {
#if __has_builtin(__builtin_amdgcn_s_wait_tensorcnt)
  __builtin_amdgcn_s_wait_tensorcnt(0);
#else
  asm volatile("s_wait_tensorcnt 0x0" ::: "memory");
#endif
}

#if USE_TDM
// 2-D tile load: rows x 32 bf16 elements from a row-major [*, kelems] matrix
// into LDS (tile stored row-major, row stride 32 elems). D# bitfields per
// CDNA5 ISA.
__device__ __forceinline__ void tdm_load_tile_2d(unsigned lds_addr, const void* gaddr,
                                                 int rows, int kelems) {
  unsigned long ga = (unsigned long)gaddr;
  v4u g0;
  g0.x = 1u;                                                  // count=1, user D#
  g0.y = lds_addr;                                            // lds_addr[31:0]
  g0.z = (unsigned)ga;                                        // global_addr[31:0]
  g0.w = (unsigned)((ga >> 32) & 0x01ffffffu) | (2u << 30);   // addr[56:32] | type=2
  v8i g1;
  g1[0] = 0x00010000;                                         // data_size=1 (2B)
  g1[1] = (int)(((unsigned)kelems & 0xffffu) << 16);          // tensor_dim0[15:0]
  g1[2] = (int)((((unsigned)kelems >> 16) & 0xffffu) |        // tensor_dim0[31:16]
                (((unsigned)rows & 0xffffu) << 16));          // tensor_dim1[15:0]
  g1[3] = (int)(32u << 16);                                   // tile_dim0=32 elems
  g1[4] = rows;                                               // tile_dim1=rows
  g1[5] = kelems;                                             // dim0_stride[31:0]
  g1[6] = 0;
  g1[7] = 0;
  v4i z4 = {0, 0, 0, 0};
#if defined(__clang_major__) && (__clang_major__ >= 23)
  v8i z8 = {0, 0, 0, 0, 0, 0, 0, 0};
  __builtin_amdgcn_tensor_load_to_lds(g0, g1, z4, z4, z8, 0);
#else
  __builtin_amdgcn_tensor_load_to_lds(g0, g1, z4, z4, 0);
#endif
}
#endif

// ------------------------------------------------- weight pack kernels -----
// wq/wk/wv: [L][H][D][DH] fp32 -> [L][H*DH][D] bf16  (N-major, K contiguous)
__global__ __launch_bounds__(256) void cvt_qkv_kernel(
    const float* __restrict__ w, bf16_t* __restrict__ out) {
  long o = (long)blockIdx.x * 256 + threadIdx.x;
  long total = (long)Lc * (Hn * DHc) * Dm;
  if (o >= total) return;
  int  d = (int)(o % Dm);
  int  n = (int)((o / Dm) % (Hn * DHc));
  int  l = (int)(o / ((long)Dm * Hn * DHc));
  int  h = n / DHc, e = n % DHc;
  out[o] = (bf16_t)w[(((long)(l * Hn + h) * Dm + d) * DHc) + e];
}

// generic: [L][Kd][Nd] fp32 -> [L][Nd][Kd] bf16
__global__ __launch_bounds__(256) void cvt_mat_kernel(
    const float* __restrict__ w, bf16_t* __restrict__ out,
    int Kd, int Nd, long total) {
  long o = (long)blockIdx.x * 256 + threadIdx.x;
  if (o >= total) return;
  long kk = o % Kd;
  long n  = (o / Kd) % Nd;
  long l  = o / ((long)Kd * Nd);
  out[o] = (bf16_t)w[(l * Kd + kk) * Nd + n];
}

// ----------------------------------------------------------- embedding -----
__global__ __launch_bounds__(256) void embed_kernel(
    const int* __restrict__ idx, const float* __restrict__ tok,
    const float* __restrict__ pos, float* __restrict__ x) {
  long o = (long)blockIdx.x * 256 + threadIdx.x;   // < Mrows*Dm
  int  d  = (int)(o & (Dm - 1));
  long ms = o >> 10;                               // token row (b*S + s)
  int  s  = (int)(ms & (Sc - 1));
  int  t  = idx[ms];
  x[o] = tok[(long)t * Dm + d] + pos[(long)s * Dm + d];
}

// ------------------------------------------------------------ layernorm ----
__global__ __launch_bounds__(256) void ln_kernel(
    const float* __restrict__ x, const float* __restrict__ sc,
    const float* __restrict__ bi, bf16_t* __restrict__ h) {
  __shared__ float red[256];
  const int tid = threadIdx.x;
  const float* xr = x + (long)blockIdx.x * Dm;
  float v4[4], sum = 0.f;
#pragma unroll
  for (int i = 0; i < 4; ++i) { v4[i] = xr[tid + i * 256]; sum += v4[i]; }
  red[tid] = sum; __syncthreads();
  for (int off = 128; off; off >>= 1) {
    if (tid < off) red[tid] += red[tid + off];
    __syncthreads();
  }
  float mean = red[0] * (1.0f / Dm); __syncthreads();
  float vs = 0.f;
#pragma unroll
  for (int i = 0; i < 4; ++i) { float dd = v4[i] - mean; vs += dd * dd; }
  red[tid] = vs; __syncthreads();
  for (int off = 128; off; off >>= 1) {
    if (tid < off) red[tid] += red[tid + off];
    __syncthreads();
  }
  float inv = rsqrtf(red[0] * (1.0f / Dm) + 1e-5f);
  bf16_t* hr = h + (long)blockIdx.x * Dm;
#pragma unroll
  for (int i = 0; i < 4; ++i) {
    int c = tid + i * 256;
    hr[c] = (bf16_t)((v4[i] - mean) * inv * sc[c] + bi[c]);
  }
}

// ----------------------------------------------------------- WMMA GEMM -----
// C[M,N] = A[M,K](bf16 row-major) * Wt[N,K](bf16 row-major)^T, epilogues.
// 256 thr / 8 waves; 64x128 block tile; K-step 32; double-buffered LDS.
// A tile: per-lane GLOBAL_LOAD_ASYNC_TO_LDS_B128 (ASYNCcnt).
// B tile: one TENSOR_LOAD_TO_LDS DMA issued by wave 0 (TENSORcnt).
enum { EPI_RES_F32 = 0, EPI_BIAS_RELU_BF16 = 1, EPI_QK_SCATTER = 2,
       EPI_VT_SCATTER = 3, EPI_BIAS_F32 = 4 };

__global__ __launch_bounds__(256) void gemm_bf16_kernel(
    const bf16_t* __restrict__ A, const bf16_t* __restrict__ Wt,
    const float* __restrict__ bias, const float* __restrict__ res,
    void* __restrict__ out, int M, int N, int K, int mode) {
  constexpr int TM = 64, TN = 128, TK = 32;
  __shared__ __align__(16) bf16_t lA[2][TM * TK];   // 2 x 4 KB
  __shared__ __align__(16) bf16_t lB[2][TN * TK];   // 2 x 8 KB
  const int tid = threadIdx.x, lane = tid & 31, wave = tid >> 5;
  const int wm = wave & 1, wn = wave >> 1;          // 2 x 4 wave grid
  const long m0 = (long)blockIdx.y * TM;
  const long n0 = (long)blockIdx.x * TN;

  // A staging: each thread owns one 16B chunk of the 64x32 tile
  const int srow = tid >> 2, sch = (tid & 3) * 8;
  const bf16_t* gA = A + (m0 + srow) * K + sch;
  const unsigned la = lds_addr_of(&lA[0][srow * TK + sch]);
  constexpr unsigned strA = TM * TK * 2;            // 4096 B per buffer
  constexpr unsigned strB = TN * TK * 2;            // 8192 B per buffer
#if USE_TDM
  const bf16_t* gB = Wt + n0 * K;                   // tile origin (uniform)
  const unsigned lb = lds_addr_of(&lB[0][0]);       // uniform
#else
  const bf16_t* gB0 = Wt + (n0 + srow) * K + sch;
  const bf16_t* gB1 = Wt + (n0 + 64 + srow) * K + sch;
  const unsigned lb0 = lds_addr_of(&lB[0][srow * TK + sch]);
  const unsigned lb1 = lds_addr_of(&lB[0][(64 + srow) * TK + sch]);
#endif

  auto issue_tile = [&](int buf) {
    async_copy_b128(la + (unsigned)buf * strA, gA);
    gA += TK;                                       // +64 B, next K-slab
#if USE_TDM
    if (wave == 0)                                  // single DMA per block
      tdm_load_tile_2d(lb + (unsigned)buf * strB, gB, TN, K);
    gB += TK;
#else
    async_copy_b128(lb0 + (unsigned)buf * strB, gB0);
    async_copy_b128(lb1 + (unsigned)buf * strB, gB1);
    gB0 += TK; gB1 += TK;
#endif
  };

  f32x8 acc[2][2] = {};
  const int nk = K / TK;
  issue_tile(0);                                    // prologue
  for (int kt = 0; kt < nk; ++kt) {
    const int cur = kt & 1;
    if (kt + 1 < nk) {
      issue_tile(1 - cur);                          // overlap with compute
      __builtin_prefetch(gA, 0, 1);                 // tile-after-next -> GL2
#if USE_TDM
      wait_async_le1();                             // drain current A tile
      if (wave == 0) wait_tensor_le1();             // drain current B tile
#else
      wait_async_le3();
#endif
    } else {
      wait_async_0();
#if USE_TDM
      if (wave == 0) wait_tensor_0();
#endif
    }
    __syncthreads();                                // publish LDS to all waves

    bf16x16 af[2], bfv[2];
#pragma unroll
    for (int mt = 0; mt < 2; ++mt) {   // A fragment: lane = row, contig K
      int row  = wm * 32 + mt * 16 + (lane & 15);
      int koff = (lane >> 4) ? 16 : 0;
      const bf16_t* p = &lA[cur][row * TK + koff];
      af[mt] = ld_frag_pair(p, p + 8);
    }
#pragma unroll
    for (int nt = 0; nt < 2; ++nt) {   // B fragment: lane = column, contig K
      int col  = wn * 32 + nt * 16 + (lane & 15);
      int koff = (lane >> 4) ? 16 : 0;
      bfv[nt] = ld_frag_contig(&lB[cur][col * TK + koff]);
    }
#pragma unroll
    for (int mt = 0; mt < 2; ++mt)
#pragma unroll
      for (int nt = 0; nt < 2; ++nt)
        acc[mt][nt] = wmma_bf16(af[mt], bfv[nt], acc[mt][nt]);
    __syncthreads();
  }

  // epilogue: C layout -> lane holds col (lane&15), rows r / 8+r per VGPR
#pragma unroll
  for (int mt = 0; mt < 2; ++mt)
#pragma unroll
    for (int nt = 0; nt < 2; ++nt)
#pragma unroll
      for (int r = 0; r < 8; ++r) {
        long m = m0 + wm * 32 + mt * 16 + ((lane < 16) ? r : 8 + r);
        long n = n0 + wn * 32 + nt * 16 + (lane & 15);
        float v = acc[mt][nt][r];
        if (mode == EPI_RES_F32) {
          float bb = bias ? bias[n] : 0.f;
          ((float*)out)[m * N + n] = res[m * N + n] + v + bb;
        } else if (mode == EPI_BIAS_RELU_BF16) {
          float t = v + bias[n];
          ((bf16_t*)out)[m * N + n] = (bf16_t)(t > 0.f ? t : 0.f);
        } else if (mode == EPI_QK_SCATTER) {     // [B,H,S,DH]
          long b = m >> 10, s = m & (Sc - 1), h = n >> 6, e = n & 63;
          ((bf16_t*)out)[((b * Hn + h) * Sc + s) * DHc + e] = (bf16_t)v;
        } else if (mode == EPI_VT_SCATTER) {     // [B,H,DH,S]
          long b = m >> 10, s = m & (Sc - 1), h = n >> 6, e = n & 63;
          ((bf16_t*)out)[((b * Hn + h) * DHc + e) * Sc + s] = (bf16_t)v;
        } else {                                 // EPI_BIAS_F32 (logits)
          ((float*)out)[m * N + n] = v + bias[n];
        }
      }
}

// ----------------------------------------------- fused flash attention -----
// One wave per (b,h, 16-query tile). Online softmax, key blocks of 32.
__global__ __launch_bounds__(32) void attn_kernel(
    const bf16_t* __restrict__ q, const bf16_t* __restrict__ k,
    const bf16_t* __restrict__ vT, bf16_t* __restrict__ o) {
  __shared__ __align__(16) bf16_t Pl[16 * 32];
  const int bh = blockIdx.y;                 // b*H + h
  const int q0 = blockIdx.x * 16;
  const int lane = threadIdx.x & 31;
  const long qkBase = (long)bh * Sc * DHc;   // q,k : [BH, S, DH]
  const long vBase  = (long)bh * DHc * Sc;   // vT  : [BH, DH, S]
  const float scale = 0.125f;                // DH^-0.5

  // Q fragments (A-layout) for dh chunks 0 and 32
  bf16x16 aq[2];
  {
    int row = q0 + (lane & 15);
    int koff = (lane >> 4) ? 16 : 0;
    const bf16_t* base = q + qkBase + (long)row * DHc;
    aq[0] = ld_frag_pair(base + koff,      base + 8 + koff);
    aq[1] = ld_frag_pair(base + 32 + koff, base + 40 + koff);
  }

  f32x8 oacc[4] = {};
  float mrow[8], lrow[8];
#pragma unroll
  for (int r = 0; r < 8; ++r) { mrow[r] = -3.0e38f; lrow[r] = 0.f; }

  const int nkb = (q0 + 15) / 32 + 1;        // causal key-block count
  for (int kb = 0; kb < nkb; ++kb) {
    const int kbase = kb * 32;
    // S = Q K^T for 32 keys -> two 16x16 f32 tiles
    f32x8 s0 = {}, s1 = {};
#pragma unroll
    for (int kk = 0; kk < 2; ++kk) {
      int dh0 = kk * 32;
      int koff = (lane >> 4) ? 16 : 0;
      bf16x16 b0 = ld_frag_contig(k + qkBase + (long)(kbase +      (lane & 15)) * DHc + dh0 + koff);
      bf16x16 b1 = ld_frag_contig(k + qkBase + (long)(kbase + 16 + (lane & 15)) * DHc + dh0 + koff);
      s0 = wmma_bf16(aq[kk], b0, s0);
      s1 = wmma_bf16(aq[kk], b1, s1);
    }
    const int n = lane & 15;
    float p0[8], p1[8], corr[8];
#pragma unroll
    for (int r = 0; r < 8; ++r) {            // scale + causal mask
      int rowm = (lane < 16) ? r : 8 + r;
      int qg = q0 + rowm;
      p0[r] = (kbase + n      <= qg) ? s0[r] * scale : -3.0e38f;
      p1[r] = (kbase + 16 + n <= qg) ? s1[r] * scale : -3.0e38f;
    }
#pragma unroll
    for (int r = 0; r < 8; ++r) {            // online softmax per row
      float mx = fmaxf(p0[r], p1[r]);
#pragma unroll
      for (int msk = 1; msk < 16; msk <<= 1) mx = fmaxf(mx, __shfl_xor(mx, msk, 16));
      float mnew = fmaxf(mrow[r], mx);
      corr[r] = __expf(mrow[r] - mnew);
      mrow[r] = mnew;
      p0[r] = __expf(p0[r] - mnew);
      p1[r] = __expf(p1[r] - mnew);
      float sm = p0[r] + p1[r];
#pragma unroll
      for (int msk = 1; msk < 16; msk <<= 1) sm += __shfl_xor(sm, msk, 16);
      lrow[r] = lrow[r] * corr[r] + sm;
    }
#pragma unroll
    for (int t = 0; t < 4; ++t)
#pragma unroll
      for (int r = 0; r < 8; ++r) oacc[t][r] *= corr[r];

    // C-layout P -> LDS (row-major 16x32 bf16) so it can re-enter as A-frag
#pragma unroll
    for (int r = 0; r < 8; ++r) {
      int rowm = (lane < 16) ? r : 8 + r;
      Pl[rowm * 32 + n]      = (bf16_t)p0[r];
      Pl[rowm * 32 + 16 + n] = (bf16_t)p1[r];
    }
    __syncthreads();
    bf16x16 ap;
    {
      int row = lane & 15, koff = (lane >> 4) ? 16 : 0;
      const bf16_t* pp = &Pl[row * 32 + koff];
      ap = ld_frag_pair(pp, pp + 8);
    }
#pragma unroll
    for (int t = 0; t < 4; ++t) {            // O += P V (4 dh tiles of 16)
      int koff = (lane >> 4) ? 16 : 0;
      bf16x16 bv = ld_frag_contig(vT + vBase + (long)(t * 16 + (lane & 15)) * Sc + kbase + koff);
      oacc[t] = wmma_bf16(ap, bv, oacc[t]);
    }
    __syncthreads();
  }

  // write O / l  -> o[b, s, h*DH + dh]  (heads-concat layout)
  const int b = bh / Hn, h = bh % Hn;
#pragma unroll
  for (int r = 0; r < 8; ++r) {
    int rowm = (lane < 16) ? r : 8 + r;
    float inv = 1.0f / lrow[r];
    long base = ((long)(b * Sc + q0 + rowm)) * Dm + h * DHc;
#pragma unroll
    for (int t = 0; t < 4; ++t)
      o[base + t * 16 + (lane & 15)] = (bf16_t)(oacc[t][r] * inv);
  }
}

// ---------------------------------------------------------------- host -----
extern "C" void kernel_launch(void* const* d_in, const int* in_sizes, int n_in,
                              void* d_out, int out_size, void* d_ws, size_t ws_size,
                              hipStream_t stream) {
  const int*   idx     = (const int*)  d_in[0];
  const float* tok_emb = (const float*)d_in[1];
  const float* pos_emb = (const float*)d_in[2];
  const float* ln1_s   = (const float*)d_in[3];
  const float* ln1_b   = (const float*)d_in[4];
  const float* wq      = (const float*)d_in[5];
  const float* wk      = (const float*)d_in[6];
  const float* wv      = (const float*)d_in[7];
  const float* wproj   = (const float*)d_in[8];
  const float* ln2_s   = (const float*)d_in[9];
  const float* ln2_b   = (const float*)d_in[10];
  const float* w1      = (const float*)d_in[11];
  const float* b1      = (const float*)d_in[12];
  const float* w2      = (const float*)d_in[13];
  const float* b2      = (const float*)d_in[14];
  const float* lnf_s   = (const float*)d_in[15];
  const float* lnf_b   = (const float*)d_in[16];
  const float* lm_w    = (const float*)d_in[17];
  const float* lm_b    = (const float*)d_in[18];
  float* logits = (float*)d_out;

  // ---- workspace carve-up -------------------------------------------------
  char* ws = (char*)d_ws;
  auto carve = [&](size_t bytes) {
    char* p = ws; ws += (bytes + 255) & ~(size_t)255; return p;
  };
  float*  x    = (float*) carve((size_t)Mrows * Dm * 4);
  bf16_t* h    = (bf16_t*)carve((size_t)Mrows * Dm * 2);
  bf16_t* qb   = (bf16_t*)carve((size_t)Mrows * Dm * 2);
  bf16_t* kb   = (bf16_t*)carve((size_t)Mrows * Dm * 2);
  bf16_t* vtb  = (bf16_t*)carve((size_t)Mrows * Dm * 2);
  bf16_t* ob   = (bf16_t*)carve((size_t)Mrows * Dm * 2);
  bf16_t* hid  = (bf16_t*)carve((size_t)Mrows * FF * 2);
  bf16_t* wqT  = (bf16_t*)carve((size_t)Lc * Dm * Dm * 2);
  bf16_t* wkT  = (bf16_t*)carve((size_t)Lc * Dm * Dm * 2);
  bf16_t* wvT  = (bf16_t*)carve((size_t)Lc * Dm * Dm * 2);
  bf16_t* wpT  = (bf16_t*)carve((size_t)Lc * Dm * Dm * 2);
  bf16_t* w1T  = (bf16_t*)carve((size_t)Lc * FF * Dm * 2);
  bf16_t* w2T  = (bf16_t*)carve((size_t)Lc * Dm * FF * 2);
  bf16_t* lmT  = (bf16_t*)carve((size_t)Vc * Dm * 2);

  // ---- weight packing (fp32 -> bf16, [N,K] layout) ------------------------
  {
    long tq = (long)Lc * Dm * Dm;
    int  gq = (int)((tq + 255) / 256);
    cvt_qkv_kernel<<<gq, 256, 0, stream>>>(wq, wqT);
    cvt_qkv_kernel<<<gq, 256, 0, stream>>>(wk, wkT);
    cvt_qkv_kernel<<<gq, 256, 0, stream>>>(wv, wvT);
    cvt_mat_kernel<<<gq, 256, 0, stream>>>(wproj, wpT, Dm, Dm, tq);
    long tf = (long)Lc * FF * Dm;
    int  gf = (int)((tf + 255) / 256);
    cvt_mat_kernel<<<gf, 256, 0, stream>>>(w1, w1T, Dm, FF, tf);
    cvt_mat_kernel<<<gf, 256, 0, stream>>>(w2, w2T, FF, Dm, tf);
    long tl = (long)Vc * Dm;
    cvt_mat_kernel<<<(int)((tl + 255) / 256), 256, 0, stream>>>(lm_w, lmT, Dm, Vc, tl);
  }

  // ---- embedding ----------------------------------------------------------
  embed_kernel<<<(Mrows * Dm) / 256, 256, 0, stream>>>(idx, tok_emb, pos_emb, x);

  auto gemm = [&](const bf16_t* A, const bf16_t* Wt, const float* bias,
                  const float* res, void* out, int M, int N, int K, int mode) {
    dim3 g(N / 128, M / 64);
    gemm_bf16_kernel<<<g, 256, 0, stream>>>(A, Wt, bias, res, out, M, N, K, mode);
  };

  // ---- transformer layers -------------------------------------------------
  for (int l = 0; l < Lc; ++l) {
    long wOff = (long)l * Dm * Dm;
    ln_kernel<<<Mrows, 256, 0, stream>>>(x, ln1_s + l * Dm, ln1_b + l * Dm, h);
    gemm(h, wqT + wOff, nullptr, nullptr, qb,  Mrows, Dm, Dm, EPI_QK_SCATTER);
    gemm(h, wkT + wOff, nullptr, nullptr, kb,  Mrows, Dm, Dm, EPI_QK_SCATTER);
    gemm(h, wvT + wOff, nullptr, nullptr, vtb, Mrows, Dm, Dm, EPI_VT_SCATTER);
    attn_kernel<<<dim3(Sc / 16, Bc * Hn), 32, 0, stream>>>(qb, kb, vtb, ob);
    gemm(ob, wpT + wOff, nullptr, x, x, Mrows, Dm, Dm, EPI_RES_F32);
    ln_kernel<<<Mrows, 256, 0, stream>>>(x, ln2_s + l * Dm, ln2_b + l * Dm, h);
    gemm(h,   w1T + (long)l * FF * Dm, b1 + (long)l * FF, nullptr, hid,
         Mrows, FF, Dm, EPI_BIAS_RELU_BF16);
    gemm(hid, w2T + (long)l * Dm * FF, b2 + (long)l * Dm, x, x,
         Mrows, Dm, FF, EPI_RES_F32);
  }

  // ---- final LN + LM head -------------------------------------------------
  ln_kernel<<<Mrows, 256, 0, stream>>>(x, lnf_s, lnf_b, h);
  gemm(h, lmT, lm_b, nullptr, logits, Mrows, Vc, Dm, EPI_BIAS_F32);
}
